// GraphAttentionLayer_70712341561514
// MI455X (gfx1250) — matrix operand order repaired and checked by reading, not compiled, not run on hardware
//
#include <hip/hip_runtime.h>
#include <hip/hip_bf16.h>

#define NN    8192
#define DD    64
#define ALPHA 0.2f
#define WAVES 4                 // waves per workgroup, split over the j range
#define SEG   (NN / WAVES)      // columns per wave

typedef __attribute__((ext_vector_type(16))) _Float16 v16h;
typedef __attribute__((ext_vector_type(8)))  _Float16 v8h;
typedef __attribute__((ext_vector_type(8)))  float    v8f;

// sched_group_barrier masks (LLVM AMDGPU): 0x2=VALU, 0x8=MFMA/WMMA, 0x20=VMEM read
#define SCHED_GROUP(mask, n) __builtin_amdgcn_sched_group_barrier((mask), (n), 0)

// ---------------------------------------------------------------------------
// Kernel 0: init the encoded-float atomic-max cell (monotone uint encoding).
// ---------------------------------------------------------------------------
__global__ void gat_init(unsigned* __restrict__ f2max_enc) {
  if (threadIdx.x == 0 && blockIdx.x == 0) *f2max_enc = 0u;
}

// ---------------------------------------------------------------------------
// Kernel 1: h = x @ Wt (f32), write transposed f16 copy ht[c][j];
//           f1[j] = h.a1 + b1 ; f2[j] = h.a2 + b2 ;
//           E1[j] = exp(f2) f16 ; E2[j] = exp(alpha*f2) f16 ;
//           atomic max of f2 (sign-flip uint encoding).
// ---------------------------------------------------------------------------
__global__ void gat_proj(const float* __restrict__ x,
                         const float* __restrict__ Wt,
                         const float* __restrict__ a1,
                         const float* __restrict__ b1,
                         const float* __restrict__ a2,
                         const float* __restrict__ b2,
                         float*     __restrict__ f1,
                         _Float16*  __restrict__ ht,
                         _Float16*  __restrict__ E1,
                         _Float16*  __restrict__ E2,
                         unsigned*  __restrict__ f2max_enc)
{
  __shared__ float s1[DD], s2[DD];
  const int j = blockIdx.x;
  const int c = threadIdx.x;
  const float* xr = x + j * DD;

  float acc = 0.f;
#pragma unroll
  for (int k = 0; k < DD; ++k)
    acc = fmaf(xr[k], Wt[k * DD + c], acc);

  ht[c * NN + j] = (_Float16)acc;   // transposed f16 copy for WMMA B loads
  s1[c] = acc * a1[c];
  s2[c] = acc * a2[c];
  __syncthreads();
#pragma unroll
  for (int off = DD / 2; off > 0; off >>= 1) {
    if (c < off) { s1[c] += s1[c + off]; s2[c] += s2[c + off]; }
    __syncthreads();
  }
  if (c == 0) {
    float F1 = s1[0] + b1[0];
    float F2 = s2[0] + b2[0];
    f1[j] = F1;
    E1[j] = (_Float16)__expf(F2);
    E2[j] = (_Float16)__expf(ALPHA * F2);
    unsigned ub  = __float_as_uint(F2);
    unsigned enc = (ub & 0x80000000u) ? ~ub : (ub | 0x80000000u);
    atomicMax(f2max_enc, enc);
  }
}

// ---------------------------------------------------------------------------
// Kernel 2: fused softmax(leaky_relu(f1_i + f2_j)) @ h, then ELU.
// One workgroup (4 waves) per 16-row tile; wave w covers j in [w*2048,(w+1)*2048).
// Per 32-column j-block per wave:
//   A[16x32 f16]  = max(c1*E1[j], c2*E2[j])   (exact softmax numerators)
//   B[32x16 f16]  = ht tiles, double-buffered; sched_group_barrier pins the
//                   loads-ahead-of-WMMA pipeline shape
//   5 x v_wmma_f32_16x16x32_f16 : 4 output tiles + row-sum via ones-B.
// Split-j partials reduced deterministically through LDS, then divide + ELU.
// ---------------------------------------------------------------------------
__global__ __launch_bounds__(WAVES * 32)
void gat_attend(const float*    __restrict__ f1,
                const _Float16* __restrict__ ht,
                const _Float16* __restrict__ E1,
                const _Float16* __restrict__ E2,
                const unsigned* __restrict__ f2max_enc,
                float*          __restrict__ out)
{
  __shared__ float redbuf[WAVES * 5 * 32 * 8];   // 20 KB: [wave][tile5][lane][reg]

  const int lane    = threadIdx.x & 31;
  const int wid     = threadIdx.x >> 5;
  // wave-uniform copy -> scalar (SGPR) loop bounds, no exec-masked loop
  const int wuni    = __builtin_amdgcn_readfirstlane(wid);
  const int rowbase = blockIdx.x << 4;           // 16 rows per workgroup
  const int m  = lane & 15;
  const int hi = lane >> 4;

  // decode global max(f2) from the monotone uint encoding
  const unsigned u = *f2max_enc;
  const float f2max = (u & 0x80000000u) ? __uint_as_float(u & 0x7fffffffu)
                                        : __uint_as_float(~u);

  // per-row (per-lane: A-matrix row = lane%16) softmax scalars
  const float f1v  = f1[rowbase + m];
  const float tmx  = f1v + f2max;
  const float mrow = tmx > 0.f ? tmx : ALPHA * tmx;   // lrelu(f1 + max f2)
  const _Float16 c1 = (_Float16)__expf(f1v - mrow);
  const _Float16 c2 = (_Float16)__expf(ALPHA * f1v - mrow);
  v8h c1v, c2v;
#pragma unroll
  for (int i = 0; i < 8; ++i) { c1v[i] = c1; c2v[i] = c2; }

  // A-matrix K offsets: lanes>=16 hold K=16..31; VGPRs 4..7 add K+8.
  const int kA = hi << 4;    // 0 or 16
  // B-matrix K offsets: lanes>=16 hold K=16..31.
  const int kB = hi << 4;    // 0 or 16

  const int jbeg = wuni * SEG;   // SGPR-uniform
  const int jend = jbeg + SEG;

  const _Float16* bp0 = ht + ( 0 + m) * NN + kB;
  const _Float16* bp1 = ht + (16 + m) * NN + kB;
  const _Float16* bp2 = ht + (32 + m) * NN + kB;
  const _Float16* bp3 = ht + (48 + m) * NN + kB;
  const _Float16* e1p = E1 + kA;
  const _Float16* e2p = E2 + kA;

  v8f acc0 = {}, acc1 = {}, acc2 = {}, acc3 = {}, accs = {};
  v16h ones;
#pragma unroll
  for (int i = 0; i < 16; ++i) ones[i] = (_Float16)1.0f;

  // prologue: first B tiles
  v16h B0 = *(const v16h*)(bp0 + jbeg);
  v16h B1 = *(const v16h*)(bp1 + jbeg);
  v16h B2 = *(const v16h*)(bp2 + jbeg);
  v16h B3 = *(const v16h*)(bp3 + jbeg);

  for (int j0 = jbeg; j0 < jend; j0 += 32) {
    // ---- prefetch next iteration's B tiles (consumed next iteration)
    const int jn = j0 + 32;
    const int jp = (jn < jend) ? jn : jbeg;     // safe address on last iter
    v16h nB0 = *(const v16h*)(bp0 + jp);
    v16h nB1 = *(const v16h*)(bp1 + jp);
    v16h nB2 = *(const v16h*)(bp2 + jp);
    v16h nB3 = *(const v16h*)(bp3 + jp);

    // ---- attention weights for this lane's A slice
    v8h e1lo = *(const v8h*)(e1p + j0);
    v8h e1hi = *(const v8h*)(e1p + j0 + 8);
    v8h e2lo = *(const v8h*)(e2p + j0);
    v8h e2hi = *(const v8h*)(e2p + j0 + 8);

    v8h w1lo = c1v * e1lo;   // v_pk_mul_f16
    v8h w2lo = c2v * e2lo;
    v8h w1hi = c1v * e1hi;
    v8h w2hi = c2v * e2hi;

    v16h A;
#pragma unroll
    for (int i = 0; i < 8; ++i) {
      A[i]     = __builtin_fmaxf16(w1lo[i], w2lo[i]);   // v_pk_max_num_f16
      A[8 + i] = __builtin_fmaxf16(w1hi[i], w2hi[i]);
    }

    acc0 = __builtin_amdgcn_wmma_f32_16x16x32_f16(false, A, false, B0, (short)0, acc0, false, false);
    acc1 = __builtin_amdgcn_wmma_f32_16x16x32_f16(false, A, false, B1, (short)0, acc1, false, false);
    acc2 = __builtin_amdgcn_wmma_f32_16x16x32_f16(false, A, false, B2, (short)0, acc2, false, false);
    acc3 = __builtin_amdgcn_wmma_f32_16x16x32_f16(false, A, false, B3, (short)0, acc3, false, false);
    // row-sum accumulator: D = A x 1 -> softmax denominators
    accs = __builtin_amdgcn_wmma_f32_16x16x32_f16(false, A, false, ones, (short)0, accs, false, false);

    B0 = nB0; B1 = nB1; B2 = nB2; B3 = nB3;

    // pin pipeline shape: 12 VMEM reads (prefetches) -> packed VALU -> 5 WMMA
    SCHED_GROUP(0x0020, 12);   // VMEM read group first
    SCHED_GROUP(0x0002, 16);   // pk_mul/pk_max group
    SCHED_GROUP(0x0008, 5);    // WMMA group last
  }

  // ---- deterministic split-j reduction through LDS -------------------------
  // layout: redbuf[((w*5 + t)*32 + lane)*8 + r], per-lane contiguous v8f slots
  {
    float* slot = &redbuf[((wid * 5 + 0) * 32 + lane) * 8];
    *(v8f*)(slot + 0 * 32 * 8) = acc0;
    *(v8f*)(slot + 1 * 32 * 8) = acc1;
    *(v8f*)(slot + 2 * 32 * 8) = acc2;
    *(v8f*)(slot + 3 * 32 * 8) = acc3;
    *(v8f*)(slot + 4 * 32 * 8) = accs;
  }
  __syncthreads();

  // wave w finalizes output-column tile t = w (cols w*16 .. w*16+15)
  v8f num = {}, den = {};
#pragma unroll
  for (int w = 0; w < WAVES; ++w) {
    num += *(const v8f*)&redbuf[((w * 5 + wid) * 32 + lane) * 8];
    den += *(const v8f*)&redbuf[((w * 5 + 4)   * 32 + lane) * 8];
  }

#pragma unroll
  for (int r = 0; r < 8; ++r) {
    const float inv = __builtin_amdgcn_rcpf(den[r]);
    const float v   = num[r] * inv;
    const int orow  = rowbase + (hi << 3) + r;
    out[orow * DD + (wid << 4) + m] = v > 0.f ? v : (__expf(v) - 1.f);
  }
}

// ---------------------------------------------------------------------------
extern "C" void kernel_launch(void* const* d_in, const int* in_sizes, int n_in,
                              void* d_out, int out_size, void* d_ws, size_t ws_size,
                              hipStream_t stream) {
  (void)in_sizes; (void)n_in; (void)out_size; (void)ws_size;
  const float* x  = (const float*)d_in[0];
  const float* Wt = (const float*)d_in[1];
  const float* a1 = (const float*)d_in[2];
  const float* b1 = (const float*)d_in[3];
  const float* a2 = (const float*)d_in[4];
  const float* b2 = (const float*)d_in[5];
  float* out = (float*)d_out;

  // workspace layout (~1.1 MB total)
  char* ws = (char*)d_ws;
  _Float16* ht  = (_Float16*)ws;                                   // 64*8192*2 = 1 MB
  float*    f1  = (float*)   (ws + (size_t)NN * DD * 2);           // 32 KB
  _Float16* E1  = (_Float16*)(ws + (size_t)NN * DD * 2 + NN * 4);  // 16 KB
  _Float16* E2  = (_Float16*)(ws + (size_t)NN * DD * 2 + NN * 4 + NN * 2);
  unsigned* f2m = (unsigned*)(ws + (size_t)NN * DD * 2 + NN * 4 + NN * 2 + NN * 2);

  gat_init<<<1, 1, 0, stream>>>(f2m);
  gat_proj<<<NN, DD, 0, stream>>>(x, Wt, a1, b1, a2, b2, f1, ht, E1, E2, f2m);
  // one workgroup (4 waves) per 16-row tile -> 512 blocks, 2048 waves
  gat_attend<<<NN / 16, WAVES * 32, 0, stream>>>(f1, ht, E1, E2, f2m, out);
}
